// seq2seq_luong_40647570489544
// MI455X (gfx1250) — compile-verified, matrix operand order checked by reading, not run on hardware
//
#include <hip/hip_runtime.h>

typedef __attribute__((ext_vector_type(16))) _Float16 v16h;
typedef __attribute__((ext_vector_type(8)))  float    v8f;

constexpr int kH  = 1024;   // hidden
constexpr int kV  = 100;    // vocab
constexpr int kVP = 128;    // vocab padded to 32-multiple (dual-N tiling)
constexpr int kS  = 150;    // src len
constexpr int kT  = 150;    // tgt len
constexpr int kB  = 128;    // batch
constexpr int kG  = 4096;   // 4*kH (gates)
constexpr int kL  = 4;      // encoder layers

// =====================================================================
// GEMM: C[M,N] (f32) = A[M,K] (f16, row-major) x W[N,K]^T (f16, row-major)
// 256-thread block (8 waves, wave32) covers a 128x32 stripe: each wave
// owns one 16-row M tile and TWO 16-col N tiles sharing one A fragment
// (two independent WMMA chains, half the A traffic). Fragment loads are
// contiguous v16h per lane: A and B use the same slot<->K bijection, so
// the hardware K permutation cancels. Depth-2 software pipeline: each
// WMMA pair consumes fragments loaded two pairs earlier, so load latency
// is hidden behind two WMMA issue slots. Requires K % 64 == 0 and
// K >= 64 (true for all call sites: K in {1024, 2048}).
// =====================================================================
__global__ __launch_bounds__(256) void k_gemm_wmma(
    const _Float16* __restrict__ A, const _Float16* __restrict__ W,
    float* __restrict__ C, int M, int N, int K, int accum)
{
  const int wave = threadIdx.x >> 5;
  const int lane = threadIdx.x & 31;
  const int hv   = lane >> 4;          // half of wave (0/1)
  const int r    = lane & 15;          // row within tile
  const int mBase = (blockIdx.y * 8 + wave) * 16;
  const int nBase = blockIdx.x * 32;
  const _Float16* arow  = A + (size_t)(mBase + r) * K + 16 * hv;
  const _Float16* wrow0 = W + (size_t)(nBase + r) * K + 16 * hv;
  const _Float16* wrow1 = W + (size_t)(nBase + 16 + r) * K + 16 * hv;
  v8f acc0 = {}, acc1 = {};
  if (accum) {
#pragma unroll
    for (int j = 0; j < 8; ++j) {
      acc0[j] = C[(size_t)(mBase + j + 8 * hv) * N + nBase + r];
      acc1[j] = C[(size_t)(mBase + j + 8 * hv) * N + nBase + 16 + r];
    }
  }
  // prologue: two stages in flight
  v16h a0  = *(const v16h*)(arow);
  v16h b00 = *(const v16h*)(wrow0);
  v16h b10 = *(const v16h*)(wrow1);
  v16h a1  = *(const v16h*)(arow + 32);
  v16h b01 = *(const v16h*)(wrow0 + 32);
  v16h b11 = *(const v16h*)(wrow1 + 32);
  for (int k = 64; k < K; k += 64) {
    if (k + 512 < K) {                      // uniform branch
      __builtin_prefetch(arow + k + 512, 0, 3);
      __builtin_prefetch(wrow0 + k + 512, 0, 3);
      __builtin_prefetch(wrow1 + k + 512, 0, 3);
    }
    // consume stage 0, refill with K-chunk k
    acc0 = __builtin_amdgcn_wmma_f32_16x16x32_f16(false, a0, false, b00,
                                                  (short)0, acc0, false, false);
    acc1 = __builtin_amdgcn_wmma_f32_16x16x32_f16(false, a0, false, b10,
                                                  (short)0, acc1, false, false);
    a0  = *(const v16h*)(arow + k);
    b00 = *(const v16h*)(wrow0 + k);
    b10 = *(const v16h*)(wrow1 + k);
    // consume stage 1, refill with K-chunk k+32
    acc0 = __builtin_amdgcn_wmma_f32_16x16x32_f16(false, a1, false, b01,
                                                  (short)0, acc0, false, false);
    acc1 = __builtin_amdgcn_wmma_f32_16x16x32_f16(false, a1, false, b11,
                                                  (short)0, acc1, false, false);
    a1  = *(const v16h*)(arow + k + 32);
    b01 = *(const v16h*)(wrow0 + k + 32);
    b11 = *(const v16h*)(wrow1 + k + 32);
  }
  // epilogue: drain both stages
  acc0 = __builtin_amdgcn_wmma_f32_16x16x32_f16(false, a0, false, b00,
                                                (short)0, acc0, false, false);
  acc1 = __builtin_amdgcn_wmma_f32_16x16x32_f16(false, a0, false, b10,
                                                (short)0, acc1, false, false);
  acc0 = __builtin_amdgcn_wmma_f32_16x16x32_f16(false, a1, false, b01,
                                                (short)0, acc0, false, false);
  acc1 = __builtin_amdgcn_wmma_f32_16x16x32_f16(false, a1, false, b11,
                                                (short)0, acc1, false, false);
#pragma unroll
  for (int j = 0; j < 8; ++j) {  // D layout: lane -> (m = j + 8*hv, n = r)
    C[(size_t)(mBase + j + 8 * hv) * N + nBase + r]      = acc0[j];
    C[(size_t)(mBase + j + 8 * hv) * N + nBase + 16 + r] = acc1[j];
  }
}

// ---------------------- small utility kernels ------------------------
__global__ void k_f32_to_f16(const float* __restrict__ s,
                             _Float16* __restrict__ d, int n) {
  int i = blockIdx.x * blockDim.x + threadIdx.x;
  if (i < n) d[i] = (_Float16)s[i];
}

__global__ void k_f32_to_f16_padrows(const float* __restrict__ s,
                                     _Float16* __restrict__ d,
                                     int rows, int padRows, int width) {
  int i = blockIdx.x * blockDim.x + threadIdx.x;
  if (i >= padRows * width) return;
  int row = i / width, c = i % width;
  d[i] = (row < rows) ? (_Float16)s[(size_t)row * width + c] : (_Float16)0.0f;
}

__global__ void k_zero_f32(float* d, int n) {
  int i = blockIdx.x * blockDim.x + threadIdx.x;
  if (i < n) d[i] = 0.0f;
}
__global__ void k_zero_f16(_Float16* d, int n) {
  int i = blockIdx.x * blockDim.x + threadIdx.x;
  if (i < n) d[i] = (_Float16)0.0f;
}

// embedding gather: dst[row*width + h] = tab[tok[row]*width + h]
__global__ void k_gather_rows16(const _Float16* __restrict__ tab,
                                const int* __restrict__ toks,
                                _Float16* __restrict__ dst,
                                int nrows, int width) {
  int i = blockIdx.x * blockDim.x + threadIdx.x;
  if (i >= nrows * width) return;
  int row = i / width, h = i % width;
  dst[(size_t)row * width + h] = tab[(size_t)toks[row] * width + h];
}

// dst16[b*2W + w] = a[b*W+w]; dst16[b*2W + W + w] = c[b*W+w]
__global__ void k_concat16(const _Float16* __restrict__ a,
                           const _Float16* __restrict__ c,
                           _Float16* __restrict__ dst, int W_) {
  int i = blockIdx.x * blockDim.x + threadIdx.x;
  if (i >= kB * W_) return;
  int b = i / W_, w = i % W_;
  dst[(size_t)b * 2 * W_ + w]      = a[i];
  dst[(size_t)b * 2 * W_ + W_ + w] = c[i];
}
__global__ void k_concat32to16(const float* __restrict__ a,
                               const float* __restrict__ c,
                               _Float16* __restrict__ dst, int W_) {
  int i = blockIdx.x * blockDim.x + threadIdx.x;
  if (i >= kB * W_) return;
  int b = i / W_, w = i % W_;
  dst[(size_t)b * 2 * W_ + w]      = (_Float16)a[i];
  dst[(size_t)b * 2 * W_ + W_ + w] = (_Float16)c[i];
}

// teacher-forced token embedding + context concat (f16, 128x2048)
__global__ void k_concat_emb_ctx(const _Float16* __restrict__ embD,
                                 const int* __restrict__ tgt, int t,
                                 const float* __restrict__ ctx,
                                 _Float16* __restrict__ dst) {
  int i = blockIdx.x * blockDim.x + threadIdx.x;
  if (i >= kB * kH) return;
  int b = i >> 10, h = i & (kH - 1);
  int tok = (t == 0) ? 1 : tgt[(size_t)(t - 1) * kB + b];
  dst[(size_t)b * 2 * kH + h]      = embD[(size_t)tok * kH + h];
  dst[(size_t)b * 2 * kH + kH + h] = (_Float16)ctx[i];
}

// x += bias (per column); mode 0=none 1=relu 2=tanh; optional f16 mirror
__global__ void k_bias_act(float* __restrict__ x, const float* __restrict__ bias,
                           int M, int N, int mode,
                           _Float16* __restrict__ x16, int doF16) {
  int i = blockIdx.x * blockDim.x + threadIdx.x;
  if (i >= M * N) return;
  int n = i % N;
  float v = x[i] + bias[n];
  if (mode == 1) v = fmaxf(v, 0.0f);
  else if (mode == 2) v = tanhf(v);
  x[i] = v;
  if (doF16) x16[i] = (_Float16)v;
}

__device__ __forceinline__ float sigf(float x) {
  return 1.0f / (1.0f + __expf(-x));
}

// LSTM gate nonlinearity: g (128x4096 f32, [i f g o]) + biases -> h,c
__global__ void k_lstm_pw(const float* __restrict__ g,
                          const float* __restrict__ bih,
                          const float* __restrict__ bhh,
                          float* __restrict__ c,
                          _Float16* __restrict__ h16,
                          _Float16* __restrict__ seq16, int seqStride) {
  int i = blockIdx.x * blockDim.x + threadIdx.x;
  if (i >= kB * kH) return;
  int b = i >> 10, h = i & (kH - 1);
  const float* gr = g + (size_t)b * kG;
  float gi = gr[h]          + bih[h]          + bhh[h];
  float gf = gr[kH + h]     + bih[kH + h]     + bhh[kH + h];
  float gg = gr[2 * kH + h] + bih[2 * kH + h] + bhh[2 * kH + h];
  float go = gr[3 * kH + h] + bih[3 * kH + h] + bhh[3 * kH + h];
  float c2 = sigf(gf) * c[i] + sigf(gi) * tanhf(gg);
  float h2 = sigf(go) * tanhf(c2);
  c[i] = c2;
  h16[i] = (_Float16)h2;
  if (seq16) seq16[(size_t)b * seqStride + h] = (_Float16)h2;
}

// e[b,s] = sum_h enc[s,b,h] * ah[b,h]  (one wave per (s,b) dot)
__global__ void k_att_scores(const float* __restrict__ enc,
                             const float* __restrict__ ah,
                             float* __restrict__ e) {
  int wid  = (blockIdx.x * blockDim.x + threadIdx.x) >> 5;
  int lane = threadIdx.x & 31;
  if (wid >= kS * kB) return;
  int s = wid / kB, b = wid % kB;
  const float* er = enc + ((size_t)s * kB + b) * kH;
  const float* ar = ah + (size_t)b * kH;
  float sum = 0.0f;
  for (int h = lane; h < kH; h += 32) sum += er[h] * ar[h];
  for (int o = 16; o > 0; o >>= 1) sum += __shfl_xor(sum, o, 32);
  if (lane == 0) e[(size_t)b * kS + s] = sum;
}

// in-place softmax over S per batch row
__global__ void k_softmax_row(float* __restrict__ e) {
  __shared__ float red[256];
  int b = blockIdx.x, tid = threadIdx.x;
  float* row = e + (size_t)b * kS;
  float mx = -3.4e38f;
  for (int s = tid; s < kS; s += 256) mx = fmaxf(mx, row[s]);
  red[tid] = mx; __syncthreads();
  for (int o = 128; o > 0; o >>= 1) {
    if (tid < o) red[tid] = fmaxf(red[tid], red[tid + o]);
    __syncthreads();
  }
  mx = red[0]; __syncthreads();
  float sum = 0.0f;
  for (int s = tid; s < kS; s += 256) {
    float ex = __expf(row[s] - mx);
    row[s] = ex; sum += ex;
  }
  red[tid] = sum; __syncthreads();
  for (int o = 128; o > 0; o >>= 1) {
    if (tid < o) red[tid] += red[tid + o];
    __syncthreads();
  }
  float inv = 1.0f / red[0];
  for (int s = tid; s < kS; s += 256) row[s] *= inv;
}

// ctx[b,h] = sum_s enc[s,b,h]*att[b,s]
__global__ void k_att_ctx(const float* __restrict__ enc,
                          const float* __restrict__ att,
                          float* __restrict__ ctx) {
  int i = blockIdx.x * blockDim.x + threadIdx.x;
  if (i >= kB * kH) return;
  int b = i >> 10, h = i & (kH - 1);
  float sum = 0.0f;
  for (int s = 0; s < kS; ++s)
    sum += enc[((size_t)s * kB + b) * kH + h] * att[(size_t)b * kS + s];
  ctx[i] = sum;
}

// per-step output head: softmax over V, argmax, -log p(target)
__global__ void k_out_step(const float* __restrict__ logits,
                           const float* __restrict__ outb,
                           const int* __restrict__ tgt, int t,
                           float* __restrict__ inference,
                           float* __restrict__ correct,
                           float* __restrict__ logp) {
  int b = threadIdx.x;
  if (b >= kB) return;
  const float* lr = logits + (size_t)b * kVP;
  float mx = -3.4e38f; int am = 0;
  for (int v = 0; v < kV; ++v) {
    float l = lr[v] + outb[v];
    if (l > mx) { mx = l; am = v; }
  }
  float sum = 0.0f;
  for (int v = 0; v < kV; ++v) sum += __expf(lr[v] + outb[v] - mx);
  int tg = tgt[(size_t)t * kB + b];
  float p = __expf(lr[tg] + outb[tg] - mx) / sum;
  inference[(size_t)t * kB + b] = (float)am;
  correct[(size_t)t * kB + b]   = (am == tg) ? 1.0f : 0.0f;
  logp[(size_t)t * kB + b]      = -__logf(fmaxf(p, 1e-10f));
}

// loss / acc / all_acc (deterministic tree reductions)
__global__ void k_finalize(const float* __restrict__ correct,
                           const float* __restrict__ logp,
                           float* __restrict__ tail) {
  __shared__ float red[256];
  int tid = threadIdx.x;
  float s = 0.0f;
  for (int i = tid; i < kT * kB; i += 256) s += correct[i];
  red[tid] = s; __syncthreads();
  for (int o = 128; o > 0; o >>= 1) { if (tid < o) red[tid] += red[tid + o]; __syncthreads(); }
  float sumAll = red[0]; __syncthreads();
  float sl = 0.0f;
  for (int i = tid; i < kT * kB; i += 256) sl += logp[i];
  red[tid] = sl; __syncthreads();
  for (int o = 128; o > 0; o >>= 1) { if (tid < o) red[tid] += red[tid + o]; __syncthreads(); }
  float sumLoss = red[0]; __syncthreads();
  float cols = 0.0f;
  for (int b = tid; b < kB; b += 256) {
    float bad = 0.0f;
    for (int t = 0; t < kT; ++t) bad += 1.0f - correct[(size_t)t * kB + b];
    cols += (bad == 0.0f) ? 1.0f : 0.0f;
  }
  red[tid] = cols; __syncthreads();
  for (int o = 128; o > 0; o >>= 1) { if (tid < o) red[tid] += red[tid + o]; __syncthreads(); }
  if (tid == 0) {
    tail[0] = sumLoss / (float)(kT * kB);   // loss
    tail[1] = red[0] / (float)kB;           // acc
    tail[2] = sumAll / (float)(kT * kB);    // all_acc
  }
}

// =====================================================================
extern "C" void kernel_launch(void* const* d_in, const int* in_sizes, int n_in,
                              void* d_out, int out_size, void* d_ws, size_t ws_size,
                              hipStream_t stream) {
  (void)in_sizes; (void)n_in; (void)out_size; (void)ws_size;

  // ---- input unpacking (setup_inputs dict order) ----
  const int* input_seq  = (const int*)d_in[0];
  const int* target_seq = (const int*)d_in[1];
  auto encP = [&](int l, int dir, int w) -> const float* {
    return (const float*)d_in[2 + 8 * l + 4 * dir + w];  // w: 0=Wih 1=Whh 2=bih 3=bhh
  };
  const float* embed_enc = (const float*)d_in[34];
  const float* embed_dec = (const float*)d_in[35];
  const float* attW  = (const float*)d_in[36];
  const float* attb  = (const float*)d_in[37];
  const float* mlp2W = (const float*)d_in[38];
  const float* mlp2b = (const float*)d_in[39];
  auto decP = [&](int cell, int w) -> const float* {
    return (const float*)d_in[40 + 4 * cell + w];
  };
  const float* outW = (const float*)d_in[56];
  const float* outb = (const float*)d_in[57];
  float* out = (float*)d_out;

  // ---- workspace carving (256B bump allocator) ----
  char* wp = (char*)d_ws; size_t off = 0;
  auto take = [&](size_t bytes) -> void* {
    void* r = wp + off;
    off = (off + bytes + 255) & ~(size_t)255;
    return r;
  };
  _Float16 *eWih16[kL][2], *eWhh16[kL][2];
  for (int l = 0; l < kL; ++l) {
    int inK = (l == 0) ? kH : 2 * kH;
    for (int d = 0; d < 2; ++d) {
      eWih16[l][d] = (_Float16*)take((size_t)kG * inK * 2);
      eWhh16[l][d] = (_Float16*)take((size_t)kG * kH * 2);
    }
  }
  _Float16 *dWih16[4], *dWhh16[4];
  for (int c = 0; c < 4; ++c) {
    dWih16[c] = (_Float16*)take((size_t)kG * kH * 2);
    dWhh16[c] = (_Float16*)take((size_t)kG * kH * 2);
  }
  _Float16* attW16  = (_Float16*)take((size_t)kH * kH * 2);
  _Float16* mlp2W16 = (_Float16*)take((size_t)kH * 2 * kH * 2);
  _Float16* outW16  = (_Float16*)take((size_t)kVP * kH * 2);
  _Float16* embE16  = (_Float16*)take((size_t)kV * kH * 2);
  _Float16* embD16  = (_Float16*)take((size_t)kV * kH * 2);
  _Float16* xA      = (_Float16*)take((size_t)kS * kB * 2 * kH * 2);
  _Float16* xB      = (_Float16*)take((size_t)kS * kB * 2 * kH * 2);
  float*    enc_out = (float*)take((size_t)kS * kB * kH * 4);
  float*    g       = (float*)take((size_t)kB * kG * 4);
  _Float16* h16[2]  = {(_Float16*)take((size_t)kB * kH * 2),
                       (_Float16*)take((size_t)kB * kH * 2)};
  float*    c32[2]  = {(float*)take((size_t)kB * kH * 4),
                       (float*)take((size_t)kB * kH * 4)};
  _Float16* hfb16   = (_Float16*)take((size_t)kB * 2 * kH * 2);
  _Float16* cfb16   = (_Float16*)take((size_t)kB * 2 * kH * 2);
  float*    cD[4];  _Float16* h16D[4];
  for (int c = 0; c < 4; ++c) {
    cD[c]   = (float*)take((size_t)kB * kH * 4);
    h16D[c] = (_Float16*)take((size_t)kB * kH * 2);
  }
  float*    ah32   = (float*)take((size_t)kB * kH * 4);
  float*    tmp32  = (float*)take((size_t)kB * kH * 4);
  float*    ctx32  = (float*)take((size_t)kB * kH * 4);
  float*    e32    = (float*)take((size_t)kB * kS * 4);
  float*    logits = (float*)take((size_t)kB * kVP * 4);
  _Float16* cat16  = (_Float16*)take((size_t)kB * 2 * kH * 2);
  _Float16* xin16  = (_Float16*)take((size_t)kB * kH * 2);
  float*    correct = (float*)take((size_t)kT * kB * 4);
  float*    logp    = (float*)take((size_t)kT * kB * 4);

  // ---- helpers ----
  auto conv = [&](const float* s, _Float16* d, int n) {
    k_f32_to_f16<<<(n + 255) / 256, 256, 0, stream>>>(s, d, n);
  };
  auto gemm = [&](const _Float16* A, const _Float16* W, float* C,
                  int M, int N, int K, int accum) {
    dim3 grid(N / 32, M / 128);   // dual-N tiles per wave
    k_gemm_wmma<<<grid, 256, 0, stream>>>(A, W, C, M, N, K, accum);
  };
  const int BH   = kB * kH;          // 131072
  const int gBH  = (BH + 255) / 256; // 512 blocks

  // ---- weight / embedding conversion to f16 ----
  for (int l = 0; l < kL; ++l) {
    int inK = (l == 0) ? kH : 2 * kH;
    for (int d = 0; d < 2; ++d) {
      conv(encP(l, d, 0), eWih16[l][d], kG * inK);
      conv(encP(l, d, 1), eWhh16[l][d], kG * kH);
    }
  }
  for (int c = 0; c < 4; ++c) {
    conv(decP(c, 0), dWih16[c], kG * kH);
    conv(decP(c, 1), dWhh16[c], kG * kH);
  }
  conv(attW, attW16, kH * kH);
  conv(mlp2W, mlp2W16, kH * 2 * kH);
  conv(embed_enc, embE16, kV * kH);
  conv(embed_dec, embD16, kV * kH);
  k_f32_to_f16_padrows<<<(kVP * kH + 255) / 256, 256, 0, stream>>>(
      outW, outW16, kV, kVP, kH);

  // ---- encoder ----
  {
    int n = kS * kB * kH;  // embed gather into xA (stride kH for layer 0)
    k_gather_rows16<<<(n + 255) / 256, 256, 0, stream>>>(
        embE16, input_seq, xA, kS * kB, kH);
  }
  _Float16* cur = xA;
  _Float16* nxt = xB;
  for (int l = 0; l < kL; ++l) {
    int inK = (l == 0) ? kH : 2 * kH;
    for (int dir = 0; dir < 2; ++dir) {
      k_zero_f16<<<gBH, 256, 0, stream>>>(h16[dir], BH);
      k_zero_f32<<<gBH, 256, 0, stream>>>(c32[dir], BH);
      for (int step = 0; step < kS; ++step) {
        int t = (dir == 0) ? step : (kS - 1 - step);
        gemm(cur + (size_t)t * kB * inK, eWih16[l][dir], g, kB, kG, inK, 0);
        gemm(h16[dir], eWhh16[l][dir], g, kB, kG, kH, 1);
        k_lstm_pw<<<gBH, 256, 0, stream>>>(
            g, encP(l, dir, 2), encP(l, dir, 3), c32[dir], h16[dir],
            nxt + (size_t)t * kB * 2 * kH + dir * kH, 2 * kH);
      }
    }
    _Float16* tmp = cur; cur = nxt; nxt = tmp;
  }
  // final fwd/bwd state concat (last layer)
  k_concat16<<<gBH, 256, 0, stream>>>(h16[0], h16[1], hfb16, kH);
  k_concat32to16<<<gBH, 256, 0, stream>>>(c32[0], c32[1], cfb16, kH);

  // enc_out = x @ mlp2W^T + mlp2b   (19200 x 1024, K=2048)
  gemm(cur, mlp2W16, enc_out, kS * kB, kH, 2 * kH, 0);
  k_bias_act<<<(kS * kB * kH + 255) / 256, 256, 0, stream>>>(
      enc_out, mlp2b, kS * kB, kH, 0, (_Float16*)nullptr, 0);

  // decoder initial states
  gemm(hfb16, mlp2W16, tmp32, kB, kH, 2 * kH, 0);
  k_bias_act<<<gBH, 256, 0, stream>>>(tmp32, mlp2b, kB, kH, 0, h16D[0], 1);
  gemm(cfb16, mlp2W16, cD[0], kB, kH, 2 * kH, 0);
  k_bias_act<<<gBH, 256, 0, stream>>>(cD[0], mlp2b, kB, kH, 0, (_Float16*)nullptr, 0);
  for (int c = 1; c < 4; ++c) {
    k_zero_f32<<<gBH, 256, 0, stream>>>(cD[c], BH);
    k_zero_f16<<<gBH, 256, 0, stream>>>(h16D[c], BH);
  }

  // ---- decoder loop ----
  for (int t = 0; t < kT; ++t) {
    // attention
    gemm(h16D[0], attW16, ah32, kB, kH, kH, 0);
    k_bias_act<<<gBH, 256, 0, stream>>>(ah32, attb, kB, kH, 1, (_Float16*)nullptr, 0);
    k_att_scores<<<(kS * kB * 32) / 256, 256, 0, stream>>>(enc_out, ah32, e32);
    k_softmax_row<<<kB, 256, 0, stream>>>(e32);
    k_att_ctx<<<gBH, 256, 0, stream>>>(enc_out, e32, ctx32);
    // xin = tanh([emb, ctx] @ mlp2W^T + mlp2b)
    k_concat_emb_ctx<<<gBH, 256, 0, stream>>>(embD16, target_seq, t, ctx32, cat16);
    gemm(cat16, mlp2W16, tmp32, kB, kH, 2 * kH, 0);
    k_bias_act<<<gBH, 256, 0, stream>>>(tmp32, mlp2b, kB, kH, 2, xin16, 1);
    // 4 stacked LSTM cells
    const _Float16* xin = xin16;
    for (int c = 0; c < 4; ++c) {
      gemm(xin, dWih16[c], g, kB, kG, kH, 0);
      gemm(h16D[c], dWhh16[c], g, kB, kG, kH, 1);
      k_lstm_pw<<<gBH, 256, 0, stream>>>(
          g, decP(c, 2), decP(c, 3), cD[c], h16D[c], (_Float16*)nullptr, 0);
      xin = h16D[c];
    }
    // output head (N padded to 128)
    gemm(h16D[3], outW16, logits, kB, kVP, kH, 0);
    k_out_step<<<1, 128, 0, stream>>>(logits, outb, target_seq, t,
                                      out, correct, logp);
  }

  k_finalize<<<1, 256, 0, stream>>>(correct, logp, out + (size_t)kT * kB);
}
